// GCN_44306882625626
// MI455X (gfx1250) — matrix operand order repaired and checked by reading, not compiled, not run on hardware
//
#include <hip/hip_runtime.h>

// ---------------------------------------------------------------------------
// 2-layer GCN forward for MI455X (gfx1250, wave32, WMMA bf16).
// out[1,10] = sum_i log_softmax( GCN2( relu( GCN1(x) ) ) )[i] / N
// ---------------------------------------------------------------------------

typedef __bf16 bf16;
typedef __attribute__((ext_vector_type(16))) __bf16 v16bf;
typedef __attribute__((ext_vector_type(8)))  float  v8f;

#define IN_F 64
#define HID  128
#define CLS  10
#define CLSP 16   // padded classes for 16x16 WMMA tiles

struct __align__(8) bf16x4 { bf16 a, b, c, d; };

static __device__ __forceinline__ bf16 f2bf(float f) {
    unsigned u = __builtin_bit_cast(unsigned, f);
    unsigned r = u + 0x7FFFu + ((u >> 16) & 1u);   // RNE
    unsigned short h = (unsigned short)(r >> 16);
    return __builtin_bit_cast(bf16, h);
}

// ---------------- init / conversion kernels ----------------

__global__ void k_zero(int* __restrict__ deg, float* __restrict__ out, int n) {
    int i = blockIdx.x * blockDim.x + threadIdx.x;
    if (i < n) deg[i] = 0;
    if (i < CLS) out[i] = 0.0f;
}

// x -> bf16, vectorized (total4 = N*IN_F/4)
__global__ void k_cvt_x(const float* __restrict__ x, bf16* __restrict__ xb, long total4) {
    long t = (long)blockIdx.x * blockDim.x + threadIdx.x;
    if (t < total4) {
        float4 v = ((const float4*)x)[t];
        bf16x4 o = { f2bf(v.x), f2bf(v.y), f2bf(v.z), f2bf(v.w) };
        ((bf16x4*)xb)[t] = o;
    }
}

// w1t[col*64 + k] = W1[k*128 + col]   (transposed, [HID][IN_F])
__global__ void k_cvt_w1(const float* __restrict__ W1, bf16* __restrict__ w1t) {
    int i = blockIdx.x * blockDim.x + threadIdx.x;   // i = k*128+col
    if (i < IN_F * HID) {
        int k = i >> 7, col = i & 127;
        w1t[col * IN_F + k] = f2bf(W1[i]);
    }
}

// w2t[col*128 + k] = (col<10) ? W2[k*10 + col] : 0   ([CLSP][HID])
__global__ void k_cvt_w2(const float* __restrict__ W2, bf16* __restrict__ w2t) {
    int i = blockIdx.x * blockDim.x + threadIdx.x;
    if (i < CLSP * HID) {
        int col = i >> 7, k = i & 127;
        w2t[i] = (col < CLS) ? f2bf(W2[k * CLS + col]) : f2bf(0.0f);
    }
}

// ---------------- degree / normalization ----------------

__global__ void k_deg(const int* __restrict__ colp, int* __restrict__ deg, long E) {
    long e = (long)blockIdx.x * blockDim.x + threadIdx.x;
    if (e < E) atomicAdd(&deg[colp[e]], 1);
}

__global__ void k_dis(const int* __restrict__ deg, float* __restrict__ dis, int n) {
    int i = blockIdx.x * blockDim.x + threadIdx.x;
    if (i < n) dis[i] = rsqrtf((float)deg[i] + 1.0f);   // +1 self loop
}

// ---------------- GEMM1: h1[N,128] = xb[N,64] @ W1  (bf16 WMMA) ----------------

__global__ __launch_bounds__(128) void k_gemm1(const bf16* __restrict__ xb,
                                               const bf16* __restrict__ w1t,
                                               float* __restrict__ h1, int n) {
    __shared__ __align__(16) bf16 lw[HID * IN_F];    // transposed W1: [col][k], 16 KB
    int tid = threadIdx.x;
    {   // cooperative 16B-wide staging: 1024 uint4, 8 per thread
        const uint4* src = (const uint4*)w1t;
        uint4* dst = (uint4*)lw;
#pragma unroll
        for (int i = 0; i < 8; ++i) dst[tid + i * 128] = src[tid + i * 128];
    }
    __syncthreads();

    int wave = tid >> 5, lane = tid & 31;
    int lr = lane & 15, hi = lane >> 4;           // hi: lane half
    int rowbase = (blockIdx.x * 4 + wave) * 16;
    bool full = (rowbase + 16) <= n;              // whole wave-tile in range?
    int row = rowbase + lr; if (row >= n) row = n - 1;   // clamp, keep EXEC full

    // A fragments (16x32 bf16): lane half selects K {0..7,16..23} vs {8..15,24..31}
    const bf16* ap = xb + (long)row * IN_F + hi * 8;
    v16bf a0, a1;
#pragma unroll
    for (int e = 0; e < 8; ++e) {
        a0[e]     = ap[e];        a0[e + 8] = ap[e + 16];
        a1[e]     = ap[e + 32];   a1[e + 8] = ap[e + 48];
    }

#pragma unroll
    for (int ct = 0; ct < 8; ++ct) {              // 8 column tiles of 16 -> HID=128
        int col = ct * 16 + lr;
        const bf16* bp = &lw[col * IN_F + hi * 8];    // B 32x16: lane=N, K {0..7,16..23} vs {8..15,24..31} by hi
        v16bf b0, b1;
#pragma unroll
        for (int e = 0; e < 8; ++e) { b0[e] = bp[e]; b0[e + 8] = bp[e + 16]; b1[e] = bp[e + 32]; b1[e + 8] = bp[e + 48]; }
        v8f c = {};
        c = __builtin_amdgcn_wmma_f32_16x16x32_bf16(false, a0, false, b0, (short)0, c, false, false);
        c = __builtin_amdgcn_wmma_f32_16x16x32_bf16(false, a1, false, b1, (short)0, c, false, false);
        float* op = h1 + (long)(rowbase + hi * 8) * HID + col;
        if (full) {
#pragma unroll
            for (int v = 0; v < 8; ++v) op[(long)v * HID] = c[v];
        } else {
#pragma unroll
            for (int v = 0; v < 8; ++v)
                if (rowbase + hi * 8 + v < n) op[(long)v * HID] = c[v];
        }
    }
}

// ---------------- layer-1 aggregation ----------------

// self-loop init: agg1 = dis[i]^2 * h1[i]   (vectorized, total4 = N*HID/4)
__global__ void k_init1(const float* __restrict__ h1, const float* __restrict__ dis,
                        float* __restrict__ agg1, long total4) {
    long t = (long)blockIdx.x * blockDim.x + threadIdx.x;
    if (t < total4) {
        int node = (int)(t >> 5);                 // (t*4)>>7
        float d = dis[node]; d *= d;
        float4 v = ((const float4*)h1)[t];
        float4 o = { d * v.x, d * v.y, d * v.z, d * v.w };
        ((float4*)agg1)[t] = o;
    }
}

// one thread = (edge, 4 features); total = E*32
__global__ void k_edge1(const int* __restrict__ rowp, const int* __restrict__ colp,
                        const float* __restrict__ dis, const float* __restrict__ h1,
                        float* __restrict__ agg1, long total) {
    long t = (long)blockIdx.x * blockDim.x + threadIdx.x;
    if (t >= total) return;
    long e = t >> 5; int f4 = (int)(t & 31) << 2;
    int r = rowp[e], c = colp[e];
    float nrm = dis[r] * dis[c];
    float4 hv = *(const float4*)(h1 + (long)r * HID + f4);
    float* dst = agg1 + (long)c * HID + f4;
    atomicAdd(dst + 0, hv.x * nrm);
    atomicAdd(dst + 1, hv.y * nrm);
    atomicAdd(dst + 2, hv.z * nrm);
    atomicAdd(dst + 3, hv.w * nrm);
}

// bias + relu + bf16 convert into h1's (dead) storage; total4 = N*HID/4
__global__ void k_relu(const float* __restrict__ agg1, const float* __restrict__ b1,
                       bf16* __restrict__ hb, long total4) {
    long t = (long)blockIdx.x * blockDim.x + threadIdx.x;
    if (t < total4) {
        int f4 = (int)(t & 31) << 2;
        float4 v = ((const float4*)agg1)[t];
        float4 b = *(const float4*)(b1 + f4);
        float x0 = v.x + b.x, x1 = v.y + b.y, x2 = v.z + b.z, x3 = v.w + b.w;
        bf16x4 o = { f2bf(x0 > 0.f ? x0 : 0.f), f2bf(x1 > 0.f ? x1 : 0.f),
                     f2bf(x2 > 0.f ? x2 : 0.f), f2bf(x3 > 0.f ? x3 : 0.f) };
        ((bf16x4*)hb)[t] = o;
    }
}

// ---------------- GEMM2: h2[N,16] = hb[N,128] @ W2pad  (bf16 WMMA) ----------------

__global__ __launch_bounds__(128) void k_gemm2(const bf16* __restrict__ hb,
                                               const bf16* __restrict__ w2t,
                                               float* __restrict__ h2, int n) {
    int tid = threadIdx.x;
    int wave = tid >> 5, lane = tid & 31;
    int lr = lane & 15, hi = lane >> 4;
    int rowbase = (blockIdx.x * 4 + wave) * 16;
    bool full = (rowbase + 16) <= n;
    int row = rowbase + lr; if (row >= n) row = n - 1;

    const bf16* ap = hb  + (long)row * HID + hi * 8;
    const bf16* bp = w2t + lr * HID + hi * 8;     // [col][k], tiny: stays in cache
    v8f c = {};
#pragma unroll
    for (int ks = 0; ks < 4; ++ks) {              // K = 128 in 4 steps of 32
        v16bf a, b;
#pragma unroll
        for (int e = 0; e < 8; ++e) {
            a[e]     = ap[ks * 32 + e];
            a[e + 8] = ap[ks * 32 + 16 + e];
        }
#pragma unroll
        for (int e = 0; e < 8; ++e) { b[e] = bp[ks * 32 + e]; b[e + 8] = bp[ks * 32 + 16 + e]; }
        c = __builtin_amdgcn_wmma_f32_16x16x32_bf16(false, a, false, b, (short)0, c, false, false);
    }
    float* op = h2 + (long)(rowbase + hi * 8) * CLSP + lr;
    if (full) {
#pragma unroll
        for (int v = 0; v < 8; ++v) op[v * CLSP] = c[v];
    } else {
#pragma unroll
        for (int v = 0; v < 8; ++v)
            if (rowbase + hi * 8 + v < n) op[v * CLSP] = c[v];
    }
}

// ---------------- layer-2 aggregation ----------------

// init: agg2 = dis^2 * h2 + b2 (bias folded; pad classes 0)
__global__ void k_init2(const float* __restrict__ h2, const float* __restrict__ dis,
                        const float* __restrict__ b2, float* __restrict__ agg2, long total) {
    long i = (long)blockIdx.x * blockDim.x + threadIdx.x;
    if (i < total) {
        int node = (int)(i >> 4);
        int f = (int)(i & 15);
        float d = dis[node];
        float bv = (f < CLS) ? b2[f] : 0.0f;
        agg2[i] = d * d * h2[i] + bv;
    }
}

__global__ void k_edge2(const int* __restrict__ rowp, const int* __restrict__ colp,
                        const float* __restrict__ dis, const float* __restrict__ h2,
                        float* __restrict__ agg2, long total) {
    long t = (long)blockIdx.x * blockDim.x + threadIdx.x;
    if (t >= total) return;
    long e = t / CLS; int f = (int)(t - e * CLS);   // only the 10 real classes
    int r = rowp[e], c = colp[e];
    float nrm = dis[r] * dis[c];
    atomicAdd(&agg2[(long)c * CLSP + f], h2[(long)r * CLSP + f] * nrm);
}

// ---------------- log-softmax + global reduction ----------------

__global__ void k_final(const float* __restrict__ agg2, float* __restrict__ out, int n) {
    __shared__ float s[CLS];
    if (threadIdx.x < CLS) s[threadIdx.x] = 0.0f;
    __syncthreads();
    int i = blockIdx.x * blockDim.x + threadIdx.x;
    if (i < n) {
        float z[CLS];
        float m = -3.402823466e+38f;
#pragma unroll
        for (int c = 0; c < CLS; ++c) { z[c] = agg2[(long)i * CLSP + c]; m = fmaxf(m, z[c]); }
        float se = 0.0f;
#pragma unroll
        for (int c = 0; c < CLS; ++c) se += __expf(z[c] - m);
        float t = m + __logf(se);
#pragma unroll
        for (int c = 0; c < CLS; ++c) atomicAdd(&s[c], z[c] - t);
    }
    __syncthreads();
    if (threadIdx.x < CLS) atomicAdd(&out[threadIdx.x], s[threadIdx.x] / (float)n);
}

// ---------------------------------------------------------------------------

static inline unsigned cdiv(long a, long b) { return (unsigned)((a + b - 1) / b); }

extern "C" void kernel_launch(void* const* d_in, const int* in_sizes, int n_in,
                              void* d_out, int out_size, void* d_ws, size_t ws_size,
                              hipStream_t stream) {
    const float* x  = (const float*)d_in[0];
    const int*   ei = (const int*)  d_in[1];
    const float* W1 = (const float*)d_in[2];
    const float* b1 = (const float*)d_in[3];
    const float* W2 = (const float*)d_in[4];
    const float* b2 = (const float*)d_in[5];
    float* out = (float*)d_out;

    const int  n = in_sizes[0] / IN_F;       // 100000
    const long E = (long)in_sizes[1] / 2;    // 1.6M
    const int* rowp = ei;                    // edge_index[0]
    const int* colp = ei + E;                // edge_index[1]

    // ---- workspace carve-up (all offsets 16B aligned) ----
    char* w = (char*)d_ws;
    bf16* xb   = (bf16*)w;  w += (size_t)n * IN_F * sizeof(bf16);     // 12.8 MB
    bf16* w1t  = (bf16*)w;  w += (size_t)HID * IN_F * sizeof(bf16);   // 16 KB
    bf16* w2t  = (bf16*)w;  w += (size_t)CLSP * HID * sizeof(bf16);   // 4 KB
    float* h1  = (float*)w; w += (size_t)n * HID * sizeof(float);     // 51.2 MB
    bf16*  hb  = (bf16*)h1;                                           // alias (h1 dead by relu)
    float* agg1= (float*)w; w += (size_t)n * HID * sizeof(float);     // 51.2 MB
    float* h2  = (float*)w; w += (size_t)n * CLSP * sizeof(float);    // 6.4 MB
    float* agg2= (float*)w; w += (size_t)n * CLSP * sizeof(float);    // 6.4 MB
    int*   deg = (int*)w;   w += (size_t)n * sizeof(int);             // 0.4 MB
    float* dis = (float*)w; w += (size_t)n * sizeof(float);           // 0.4 MB

    const long nh4 = (long)n * HID / 4;    // 3.2 M  (float4 granules)
    const long nx4 = (long)n * IN_F / 4;   // 1.6 M
    const long nc  = (long)n * CLSP;       // 1.6 M
    const long e1  = E * 32;               // 51.2 M  (edge x float4)
    const long e2  = E * CLS;              // 16 M

    k_zero  <<<cdiv(n, 256), 256, 0, stream>>>(deg, out, n);
    k_cvt_x <<<cdiv(nx4, 256), 256, 0, stream>>>(x, xb, nx4);
    k_cvt_w1<<<cdiv(IN_F * HID, 256), 256, 0, stream>>>(W1, w1t);
    k_cvt_w2<<<cdiv(CLSP * HID, 256), 256, 0, stream>>>(W2, w2t);
    k_deg   <<<cdiv(E, 256), 256, 0, stream>>>(colp, deg, E);
    k_dis   <<<cdiv(n, 256), 256, 0, stream>>>(deg, dis, n);

    k_gemm1 <<<cdiv(n, 64), 128, 0, stream>>>(xb, w1t, h1, n);
    k_init1 <<<cdiv(nh4, 256), 256, 0, stream>>>(h1, dis, agg1, nh4);
    k_edge1 <<<cdiv(e1, 256), 256, 0, stream>>>(rowp, colp, dis, h1, agg1, e1);
    k_relu  <<<cdiv(nh4, 256), 256, 0, stream>>>(agg1, b1, hb, nh4);

    k_gemm2 <<<cdiv(n, 64), 128, 0, stream>>>(hb, w2t, h2, n);
    k_init2 <<<cdiv(nc, 256), 256, 0, stream>>>(h2, dis, b2, agg2, nc);
    k_edge2 <<<cdiv(e2, 256), 256, 0, stream>>>(rowp, colp, dis, h2, agg2, e2);

    k_final <<<cdiv(n, 256), 256, 0, stream>>>(agg2, out, n);
}
